// BlockLTN_34153579938696
// MI455X (gfx1250) — compile-verified, matrix operand order checked by reading, not run on hardware
//
#include <hip/hip_runtime.h>

// ---------- types ----------
typedef __attribute__((ext_vector_type(16))) __bf16      v16bf;
typedef __attribute__((ext_vector_type(2)))  __bf16      v2bf;
typedef __attribute__((ext_vector_type(8)))  float       v8f;
typedef __attribute__((ext_vector_type(4)))  unsigned int uint4v;
typedef __attribute__((ext_vector_type(4)))  int          v4i;

union FragU {
    uint4v q[2];   // two 16-byte halves (operand VGPRs 0-3, 4-7)
    v16bf  f;      // bf16 A/B fragment for v_wmma_f32_16x16x32_bf16
};

// ---------- f32 -> bf16 (native fptrunc -> v_cvt_*_bf16_f32 on gfx1250) ----------
__device__ __forceinline__ unsigned short f2bf(float x) {
    __bf16 h = (__bf16)x;
    unsigned short s;
    __builtin_memcpy(&s, &h, 2);
    return s;
}

__device__ __forceinline__ unsigned int pack2bf(float x, float y) {
#if __has_builtin(__builtin_amdgcn_cvt_pk_bf16_f32)
    auto t = __builtin_amdgcn_cvt_pk_bf16_f32(x, y);
    unsigned int u;
    __builtin_memcpy(&u, &t, 4);
    return u;
#else
    v2bf t;
    t.x = (__bf16)x;          // fptrunc: backend selects hw bf16 converter
    t.y = (__bf16)y;
    unsigned int u;
    __builtin_memcpy(&u, &t, 4);
    return u;
#endif
}

// ---------- problem dims ----------
#define DIM_V 8192
#define DIM_E 8192
#define DIM_K 64
#define DIM_C 64
#define DIM_D 8
#define DIM_O 8
#define NDIM  512          // N = O*C; also K2 = K*D for GEMM1
// ---------- tiling ----------
#define TM 128
#define TN 128
#define TK 32
#define LDSPITCH 80        // 32 bf16 (64B) + 16B pad; keeps b128 reads 16B-aligned

// ---------- CDNA5 async global->LDS (gated; safe fallback) ----------
#if __has_builtin(__builtin_amdgcn_global_load_async_to_lds_b128)
#define USE_ASYNC_LDS 1
typedef __attribute__((address_space(1))) v4i g1v4i;   // global src operand type
typedef __attribute__((address_space(3))) v4i l3v4i;   // LDS dst operand type
#else
#define USE_ASYNC_LDS 0
#endif

__device__ __forceinline__ void wait_async0() {
#if USE_ASYNC_LDS
#if __has_builtin(__builtin_amdgcn_s_wait_asynccnt)
    __builtin_amdgcn_s_wait_asynccnt(0);
#else
    asm volatile("s_wait_asynccnt 0" ::: "memory");
#endif
#endif
}

// =====================================================================
// Prep: Wrt[n][kk] = bf16(W[o,d,k,c]),  kk = k*8+d, n = o*64+c  (transposed!)
// =====================================================================
__global__ void prep_w_kernel(const float* __restrict__ W, unsigned short* __restrict__ Wrt) {
    int idx = blockIdx.x * 256 + threadIdx.x;   // nn*512 + kk
    int nn = idx >> 9;
    int kk = idx & (NDIM - 1);
    int k = kk >> 3, d = kk & 7;
    int o = nn >> 6, c = nn & 63;
    Wrt[idx] = f2bf(W[(((size_t)o * DIM_D + d) * DIM_K + k) * DIM_C + c]);
}

// bias[n] = sum_d b[o,c,d],  n = o*64+c
__global__ void prep_bias_kernel(const float* __restrict__ b, float* __restrict__ bias) {
    int nn = blockIdx.x * 256 + threadIdx.x;
    if (nn >= NDIM) return;
    int o = nn >> 6, c = nn & 63;
    float s = 0.f;
#pragma unroll
    for (int d = 0; d < DIM_D; ++d) s += b[((size_t)o * DIM_C + c) * DIM_D + d];
    bias[nn] = s;
}

// ---------- fill helpers ----------
__device__ __forceinline__ void issue_loadA(const float* __restrict__ A, int lda,
                                            int m_tile, int k0, int tid, float4 av[4]) {
#pragma unroll
    for (int i = 0; i < 4; ++i) {
        int chunk = tid + i * 256;               // 0..1023 (4 floats each)
        int row   = chunk >> 3;                  // 0..127
        int col   = (chunk & 7) << 2;            // 0..28
        av[i] = *(const float4*)(A + (size_t)(m_tile + row) * lda + k0 + col);
    }
}

__device__ __forceinline__ void storeA_lds(unsigned char* __restrict__ buf, int tid,
                                           const float4 av[4]) {
#pragma unroll
    for (int i = 0; i < 4; ++i) {
        int chunk = tid + i * 256;
        int row   = chunk >> 3;
        int col   = (chunk & 7) << 2;
        unsigned int p0 = pack2bf(av[i].x, av[i].y);
        unsigned int p1 = pack2bf(av[i].z, av[i].w);
        *(uint2*)(buf + row * LDSPITCH + col * 2) = make_uint2(p0, p1);
    }
}

// B is pre-transposed: Bt[n][k], row stride == Ktot. Tile: 128 rows x 64B.
__device__ __forceinline__ void fillB(const unsigned short* __restrict__ Bt, int Ktot,
                                      int n_tile, int k0, int tid,
                                      unsigned char* __restrict__ buf) {
    int row = tid >> 1;                          // 0..127
    int seg = tid & 1;                           // two 32B halves of the 64B row
    const unsigned short* g = Bt + (size_t)(n_tile + row) * Ktot + k0 + seg * 16;
    unsigned char*        l = buf + row * LDSPITCH + seg * 32;
#if USE_ASYNC_LDS
    __builtin_amdgcn_global_load_async_to_lds_b128((g1v4i*)g,       (l3v4i*)l,        0, 0);
    __builtin_amdgcn_global_load_async_to_lds_b128((g1v4i*)(g + 8), (l3v4i*)(l + 16), 0, 0);
#else
    uint4v b0 = *(const uint4v*)g;
    uint4v b1 = *(const uint4v*)(g + 8);
    *(uint4v*)l        = b0;
    *(uint4v*)(l + 16) = b1;
#endif
}

// ---------- one K-step of WMMAs from an LDS buffer pair ----------
__device__ __forceinline__ void compute_step(const unsigned char* __restrict__ bufA,
                                             const unsigned char* __restrict__ bufB,
                                             int wave_m, int wave_n, int lm, int half,
                                             v8f acc[4][2]) {
    FragU aF[4], bF[2];
#pragma unroll
    for (int ms = 0; ms < 4; ++ms) {
        const unsigned char* rp = bufA + (wave_m * 64 + ms * 16 + lm) * LDSPITCH;
        aF[ms].q[0] = *(const uint4v*)(rp + 16 * half);
        aF[ms].q[1] = *(const uint4v*)(rp + 32 + 16 * half);
    }
#pragma unroll
    for (int ns = 0; ns < 2; ++ns) {
        const unsigned char* rp = bufB + (wave_n * 32 + ns * 16 + lm) * LDSPITCH;
        bF[ns].q[0] = *(const uint4v*)(rp + 16 * half);
        bF[ns].q[1] = *(const uint4v*)(rp + 32 + 16 * half);
    }
#pragma unroll
    for (int ms = 0; ms < 4; ++ms)
#pragma unroll
        for (int ns = 0; ns < 2; ++ns)
            acc[ms][ns] = __builtin_amdgcn_wmma_f32_16x16x32_bf16(
                false, aF[ms].f, false, bF[ns].f,
                (short)0, acc[ms][ns], false, false);
}

// =====================================================================
// bf16 WMMA GEMM:  C[M x 512] = A_f32[M x Ktot] * Bt_bf16[512 x Ktot]^T
// MODE 0: z_t[n][m] bf16 = result + bias[n]   (one b128 store / subtile)
// MODE 1: out[m*512 + (n&63)*8 + (n>>6)] f32  ([E,C,O] permute)
// 256 threads = 8 waves in 2x4; each wave: 4x2 16x16 subtiles; double-buffered.
// =====================================================================
template <int MODE>
__global__ __launch_bounds__(256)
void wmma_gemm_kernel(const float* __restrict__ A,
                      const unsigned short* __restrict__ Bt,
                      const float* __restrict__ bias,
                      void* __restrict__ Cout,
                      int Ktot, int lda) {
    __shared__ __align__(16) unsigned char smA[2][TM * LDSPITCH];
    __shared__ __align__(16) unsigned char smB[2][TN * LDSPITCH];

    const int tid    = threadIdx.x;
    const int lane   = tid & 31;
    const int wid    = tid >> 5;
    const int wave_m = wid >> 2;        // 0..1 -> 64 rows
    const int wave_n = wid & 3;         // 0..3 -> 32 cols
    const int lm     = lane & 15;
    const int half   = lane >> 4;

    const int n_tile = blockIdx.x * TN;   // x = N-tile: same-M blocks adjacent -> G hits L2
    const int m_tile = blockIdx.y * TM;

    v8f acc[4][2];
#pragma unroll
    for (int ms = 0; ms < 4; ++ms)
#pragma unroll
        for (int ns = 0; ns < 2; ++ns)
            acc[ms][ns] = (v8f){0.f, 0.f, 0.f, 0.f, 0.f, 0.f, 0.f, 0.f};

    const int nsteps = Ktot / TK;

    // ---- prologue: fill buffer 0 ----
    {
        float4 av[4];
        issue_loadA(A, lda, m_tile, 0, tid, av);
        fillB(Bt, Ktot, n_tile, 0, tid, smB[0]);
        storeA_lds(smA[0], tid, av);
    }

    // ---- steady state: branch-free pipeline, one barrier per K-step ----
    int cur = 0;
    for (int s = 0; s < nsteps - 1; ++s) {
        wait_async0();
        __syncthreads();
        const int nxt = cur ^ 1;
        const int k0n = (s + 1) * TK;
        float4 av[4];
        issue_loadA(A, lda, m_tile, k0n, tid, av);    // global loads fly over WMMAs
        fillB(Bt, Ktot, n_tile, k0n, tid, smB[nxt]);  // async DMA into other buffer
        compute_step(smA[cur], smB[cur], wave_m, wave_n, lm, half, acc);
        storeA_lds(smA[nxt], tid, av);                // convert after WMMAs issued
        cur = nxt;
    }
    // ---- final K-step (no new fills) ----
    wait_async0();
    __syncthreads();
    compute_step(smA[cur], smB[cur], wave_m, wave_n, lm, half, acc);

    // ---- epilogue ----
#pragma unroll
    for (int ms = 0; ms < 4; ++ms) {
#pragma unroll
        for (int ns = 0; ns < 2; ++ns) {
            const int ng = n_tile + wave_n * 32 + ns * 16 + lm;
            if (MODE == 0) {
                // z_t[n][m]: VGPR r -> m = base + r + 8*half => 8 contiguous bf16
                const float bb = bias[ng];
                unsigned int w[4];
#pragma unroll
                for (int r = 0; r < 4; ++r)
                    w[r] = pack2bf(acc[ms][ns][2 * r] + bb, acc[ms][ns][2 * r + 1] + bb);
                const int mg = m_tile + wave_m * 64 + ms * 16 + 8 * half;
                uint4v pk = {w[0], w[1], w[2], w[3]};
                *(uint4v*)((unsigned short*)Cout + (size_t)ng * DIM_V + mg) = pk;
            } else {
                const int c = ng & 63, o = ng >> 6;
#pragma unroll
                for (int r = 0; r < 8; ++r) {
                    const int mg = m_tile + wave_m * 64 + ms * 16 + r + 8 * half;
                    ((float*)Cout)[(size_t)mg * NDIM + c * DIM_O + o] = acc[ms][ns][r];
                }
            }
        }
    }
}

// =====================================================================
extern "C" void kernel_launch(void* const* d_in, const int* in_sizes, int n_in,
                              void* d_out, int out_size, void* d_ws, size_t ws_size,
                              hipStream_t stream) {
    const float* x = (const float*)d_in[0];   // [V, K, D]
    const float* G = (const float*)d_in[1];   // [E, V]
    const float* W = (const float*)d_in[2];   // [O, D, K, C]
    const float* b = (const float*)d_in[3];   // [O, C, D]

    unsigned char* ws = (unsigned char*)d_ws;
    unsigned short* Wrt  = (unsigned short*)ws;                       // [512][512] bf16
    float*          bias = (float*)(ws + 512 * 1024);                 // [512] f32
    unsigned short* zt   = (unsigned short*)(ws + 512 * 1024 + 4096); // [512][8192] bf16

    prep_w_kernel<<<(NDIM * NDIM) / 256, 256, 0, stream>>>(W, Wrt);
    prep_bias_kernel<<<2, 256, 0, stream>>>(b, bias);

    // GEMM1: z_t[n][v] = x[v][kk] @ Wrt[n][kk]^T + bias[n]   (M=8192, K=512)
    wmma_gemm_kernel<0><<<dim3(NDIM / TN, DIM_V / TM), 256, 0, stream>>>(
        x, Wrt, bias, (void*)zt, /*Ktot=*/NDIM, /*lda=*/NDIM);

    // GEMM2: out[e,c,o] = G[e][v] @ z_t[n][v]^T               (M=8192, K=8192)
    wmma_gemm_kernel<1><<<dim3(NDIM / TN, DIM_E / TM), 256, 0, stream>>>(
        G, zt, nullptr, d_out, /*Ktot=*/DIM_V, /*lda=*/DIM_V);
}
